// FP8Linear_2911987826909
// MI455X (gfx1250) — 8-bit matrix operand order repaired and checked by reading, compiled, not run on hardware
//
#include <hip/hip_runtime.h>
#include <hip/hip_bf16.h>
#include <stdint.h>

typedef __attribute__((ext_vector_type(16))) int   v16i;
typedef __attribute__((ext_vector_type(8)))  float v8f;
typedef int v4i_vs __attribute__((vector_size(16)));

#define MDIM 16384
#define KDIM 2048
#define NDIM 2048
#define QB   128
#define KB   (KDIM / QB)
#define BM   128
#define BN   128

#define AS1 __attribute__((address_space(1)))
#define AS3 __attribute__((address_space(3)))

#if defined(__has_builtin)
#if __has_builtin(__builtin_amdgcn_global_load_async_to_lds_b128) && \
    __has_builtin(__builtin_amdgcn_s_wait_asynccnt)
#define USE_ASYNC_LDS 1
#endif
#endif

__device__ inline uint32_t f32_to_e4m3(float x) {
  uint32_t sgn = (__float_as_uint(x) >> 31) << 7;
  float a = __builtin_fabsf(x);
  if (a != a) return sgn | 0x7Fu;
  if (a >= 448.0f) return sgn | 0x7Eu;
  if (a < 0.015625f) {
    float q = a * 512.0f;
    uint32_t qi = (uint32_t)q;
    float r = q - (float)qi;
    qi += (r > 0.5f) || (r == 0.5f && (qi & 1u));
    return sgn | qi;
  }
  uint32_t b    = __float_as_uint(a);
  uint32_t mant = (b >> 20) & 0x7u;
  uint32_t rem  = b & 0xFFFFFu;
  int e = (int)((b >> 23) & 0xFFu) - 127;
  if (rem > 0x80000u || (rem == 0x80000u && (mant & 1u))) {
    mant++;
    if (mant == 8u) { mant = 0u; e++; }
  }
  if (e > 8 || (e == 8 && mant > 6u)) return sgn | 0x7Eu;
  return sgn | ((uint32_t)(e + 7) << 3) | mant;
}

__global__ __launch_bounds__(256)
void quant_x_kernel(const float* __restrict__ x,
                    unsigned char* __restrict__ xq,
                    float* __restrict__ xs  ) {
  const int wid  = threadIdx.x >> 5;
  const int lane = threadIdx.x & 31;
  const long long blk = (long long)blockIdx.x * 8 + wid;
  const int m  = (int)(blk / KB);
  const int kb = (int)(blk % KB);

  const float4 v = *(const float4*)(x + (size_t)m * KDIM + kb * QB + lane * 4);
  float f0 = __bfloat162float(__float2bfloat16(v.x));
  float f1 = __bfloat162float(__float2bfloat16(v.y));
  float f2 = __bfloat162float(__float2bfloat16(v.z));
  float f3 = __bfloat162float(__float2bfloat16(v.w));

  float amax = fmaxf(fmaxf(__builtin_fabsf(f0), __builtin_fabsf(f1)),
                     fmaxf(__builtin_fabsf(f2), __builtin_fabsf(f3)));
  #pragma unroll
  for (int o = 16; o; o >>= 1) amax = fmaxf(amax, __shfl_xor(amax, o, 32));

  float s = fmaxf(amax, 1e-4f) * (1.0f / 448.0f);
  int e; float fr = frexpf(s, &e);
  int ce = (fr == 0.5f) ? (e - 1) : e;
  float scale = ldexpf(1.0f, ce);
  float inv   = ldexpf(1.0f, -ce);

  uint32_t p =  f32_to_e4m3(f0 * inv)
             | (f32_to_e4m3(f1 * inv) << 8)
             | (f32_to_e4m3(f2 * inv) << 16)
             | (f32_to_e4m3(f3 * inv) << 24);
  *(uint32_t*)(xq + (size_t)m * KDIM + kb * QB + lane * 4) = p;
  if (lane == 0) xs[(size_t)kb * MDIM + m] = scale;
}

__global__ __launch_bounds__(256)
void quant_w_kernel(const float* __restrict__ w, unsigned char* __restrict__ wq) {
  size_t t = (size_t)blockIdx.x * 256 + threadIdx.x;
  float4 v = ((const float4*)w)[t];
  uint32_t p =  f32_to_e4m3(v.x)
             | (f32_to_e4m3(v.y) << 8)
             | (f32_to_e4m3(v.z) << 16)
             | (f32_to_e4m3(v.w) << 24);
  ((uint32_t*)wq)[t] = p;
}

__global__ __launch_bounds__(256)
void gemm_fp8_kernel(const unsigned char* __restrict__ xq,
                     const float* __restrict__ xs,
                     const unsigned char* __restrict__ wq,
                     const float* __restrict__ wsinv,
                     __hip_bfloat16* __restrict__ out) {
  __shared__ __align__(16) unsigned char As[2][BM * QB];
  __shared__ __align__(16) unsigned char Bs[2][BN * QB];

  const int tid   = threadIdx.x;
  const int wid   = tid >> 5;
  const int lane  = tid & 31;
  const int hi    = lane >> 4;
  const int l16   = lane & 15;
  const int nBlk  = blockIdx.x * BN;
  const int mBlk  = blockIdx.y * BM;
  const int mWave = wid * 16;

  auto stage = [&](int kb, int buf) {
    const unsigned char* ga = xq + (size_t)mBlk * KDIM + (size_t)kb * QB;
    const unsigned char* gb = wq + (size_t)nBlk * KDIM + (size_t)kb * QB;
    unsigned char* sa = &As[buf][0];
    unsigned char* sb = &Bs[buf][0];
    #pragma unroll
    for (int i = 0; i < 4; ++i) {
      int idx = tid * 4 + i;
      int row = idx >> 3;
      int col = (idx & 7) * 16;
#ifdef USE_ASYNC_LDS
      __builtin_amdgcn_global_load_async_to_lds_b128(
          (AS1 v4i_vs*)(ga + (size_t)row * KDIM + col),
          (AS3 v4i_vs*)(sa + row * QB + col), 0, 0);
      __builtin_amdgcn_global_load_async_to_lds_b128(
          (AS1 v4i_vs*)(gb + (size_t)row * KDIM + col),
          (AS3 v4i_vs*)(sb + row * QB + col), 0, 0);
#else
      *(uint4*)(sa + row * QB + col) =
          *(const uint4*)(ga + (size_t)row * KDIM + col);
      *(uint4*)(sb + row * QB + col) =
          *(const uint4*)(gb + (size_t)row * KDIM + col);
#endif
    }
  };

  v8f acc[8];
  #pragma unroll
  for (int j = 0; j < 8; ++j)
    #pragma unroll
    for (int i = 0; i < 8; ++i) acc[j][i] = 0.0f;

  stage(0, 0);

  for (int kb = 0; kb < KB; ++kb) {
    const int buf = kb & 1;
#ifdef USE_ASYNC_LDS
    __builtin_amdgcn_s_wait_asynccnt(0);
#endif
    __syncthreads();
    if (kb + 1 < KB) stage(kb + 1, buf ^ 1);

    v16i a;
    {
      const unsigned char* ap = &As[buf][0] + (mWave + l16) * QB + hi * 8;
      #pragma unroll
      for (int p = 0; p < 8; ++p) {
        uint2 d = *(const uint2*)(ap + p * 16);
        a[2 * p]     = (int)d.x;
        a[2 * p + 1] = (int)d.y;
      }
    }

    const float* sp = xs + (size_t)kb * MDIM + mBlk + mWave + hi * 8;
    float4 s0 = *(const float4*)(sp);
    float4 s1 = *(const float4*)(sp + 4);
    float sw  = wsinv[blockIdx.x * KB + kb];
    float sc[8] = { s0.x * sw, s0.y * sw, s0.z * sw, s0.w * sw,
                    s1.x * sw, s1.y * sw, s1.z * sw, s1.w * sw };

    #pragma unroll
    for (int j = 0; j < 8; ++j) {
      v16i b;
      const unsigned char* bp = &Bs[buf][0] + (j * 16 + l16) * QB + hi * 8;
      #pragma unroll
      for (int g = 0; g < 8; ++g) {
        uint2 d = *(const uint2*)(bp + g * 16);
        b[2 * g]     = (int)d.x;
        b[2 * g + 1] = (int)d.y;
      }
      v8f cz = {};
      v8f c = __builtin_amdgcn_wmma_f32_16x16x128_fp8_fp8(
                  a, b, (short)0, cz, false, false);
      #pragma unroll
      for (int i = 0; i < 8; ++i) acc[j][i] += c[i] * sc[i];
    }
  }

  #pragma unroll
  for (int j = 0; j < 8; ++j) {
    int n = nBlk + j * 16 + l16;
    #pragma unroll
    for (int i = 0; i < 8; ++i) {
      int m = mBlk + mWave + hi * 8 + i;
      out[(size_t)m * NDIM + n] = __float2bfloat16(acc[j][i]);
    }
  }
}

extern "C" void kernel_launch(void* const* d_in, const int* in_sizes, int n_in,
                              void* d_out, int out_size, void* d_ws, size_t ws_size,
                              hipStream_t stream) {
  const float* x     = (const float*)d_in[0];
  const float* w     = (const float*)d_in[1];
  const float* wsinv = (const float*)d_in[2];
  __hip_bfloat16* out = (__hip_bfloat16*)d_out;

  unsigned char* xq = (unsigned char*)d_ws;
  unsigned char* wq = xq + (size_t)MDIM * KDIM;
  float*         xs = (float*)(wq + (size_t)NDIM * KDIM);

  quant_x_kernel<<<(MDIM * KB) / 8, 256, 0, stream>>>(x, xq, xs);
  quant_w_kernel<<<(size_t)NDIM * KDIM / 4 / 256, 256, 0, stream>>>(w, wq);

  dim3 grid(NDIM / BN, MDIM / BM);
  gemm_fp8_kernel<<<grid, 256, 0, stream>>>(xq, xs, wq, wsinv, out);
}
